// SegGptAttention_46308337385901
// MI455X (gfx1250) — matrix operand order repaired and checked by reading, not compiled, not run on hardware
//
#include <hip/hip_runtime.h>
#include <hip/hip_bf16.h>

// ---------------- problem constants ----------------
#define B_      4
#define Hh      56
#define Ww      28
#define NTOK    1568            // Hh*Ww
#define HID     1024
#define NHEADS  16
#define HD      64
#define BH      64              // B_*NHEADS
#define ROWS    6272            // B_*NTOK
#define SCALE   0.125f          // HD^-0.5

typedef __attribute__((ext_vector_type(16))) _Float16 v16h;
typedef __attribute__((ext_vector_type(8)))  _Float16 v8h;
typedef __attribute__((ext_vector_type(8)))  float    v8f;

// ---------------- WMMA fragment helpers (per CDNA5 ISA VGPR layouts) -------
// A 16x32 f16: lane l(0-15): row l, K k0+0..7 in elems0-7, K k0+16..23 in 8-15
//              lane l+16  : row l, K k0+8..15 / k0+24..31
__device__ __forceinline__ v16h ld_a(const _Float16* rowp, int k0, int lane) {
    const _Float16* p = rowp + k0 + ((lane >> 4) << 3);
    union { v16h v; v8h h[2]; } u;
    u.h[0] = *(const v8h*)p;
    u.h[1] = *(const v8h*)(p + 16);
    return u.v;
}
// B 32x16 f16: lane n(0-15): col n, K k0+0..7 in elems0-7, K k0+16..23 in 8-15
//              lane n+16  : K k0+8..15 / k0+24..31
// colp must point at the column's K-contiguous data (B^T row-major).
__device__ __forceinline__ v16h ld_b(const _Float16* colp, int k0, int lane) {
    const _Float16* p = colp + k0 + ((lane >> 4) << 3);
    union { v16h v; v8h h[2]; } u;
    u.h[0] = *(const v8h*)p;
    u.h[1] = *(const v8h*)(p + 16);
    return u.v;
}
__device__ __forceinline__ v8f wmma16(v16h a, v16h b, v8f c) {
    return __builtin_amdgcn_wmma_f32_16x16x32_f16(false, a, false, b,
                                                  (short)0, c, false, false);
}
// C/D 16x16 f32: element(m,n) -> lane (n + 16*(m>=8)), vgpr m&7

// 16-lane (half-group) reductions: rows of a C fragment live in 16 lanes.
__device__ __forceinline__ float grp16_max(float v) {
    #pragma unroll
    for (int m = 1; m < 16; m <<= 1) v = fmaxf(v, __shfl_xor(v, m, 32));
    return v;
}
__device__ __forceinline__ float grp16_sum(float v) {
    #pragma unroll
    for (int m = 1; m < 16; m <<= 1) v += __shfl_xor(v, m, 32);
    return v;
}

// ---------------- elementwise prep kernels ----------------
__global__ void cast_f32_f16_kernel(const float* __restrict__ in,
                                    _Float16* __restrict__ out, int n) {
    for (int i = blockIdx.x * blockDim.x + threadIdx.x; i < n;
         i += gridDim.x * blockDim.x)
        out[i] = (_Float16)in[i];
}

// out[q][k][d] = rel[(q-k+size-1)][d], f32 -> f16
__global__ void gather_rel_kernel(const float* __restrict__ rel,
                                  _Float16* __restrict__ out, int size) {
    int total = size * size * HD;
    for (int i = blockIdx.x * blockDim.x + threadIdx.x; i < total;
         i += gridDim.x * blockDim.x) {
        int d = i & (HD - 1);
        int k = (i >> 6) % size;
        int q = i / (HD * size);
        out[i] = (_Float16)rel[(q - k + size - 1) * HD + d];
    }
}

// ---------------- QKV GEMM: [6272,1024]x[3072,1024]^T + bias ----------------
// 8 waves/block, wave grid 4(M)x2(N); block tile 64x128; grid (98,24)
__global__ void __launch_bounds__(256)
qkv_gemm_kernel(const _Float16* __restrict__ Xh, const _Float16* __restrict__ Wh,
                const float* __restrict__ bias,
                _Float16* __restrict__ Qh, _Float16* __restrict__ Kh,
                _Float16* __restrict__ Vt) {
    const int lane = threadIdx.x & 31, wave = threadIdx.x >> 5;
    const int mbase = blockIdx.x * 64 + (wave & 3) * 16;
    const int nbase = blockIdx.y * 128 + (wave >> 2) * 64;
    const int n_ = lane & 15, m0 = (lane >> 4) << 3;

    const _Float16* arow = Xh + (size_t)(mbase + (lane & 15)) * HID;
    v8f c0 = {}, c1 = {}, c2 = {}, c3 = {};
    for (int k0 = 0; k0 < HID; k0 += 32) {
        v16h a = ld_a(arow, k0, lane);
        c0 = wmma16(a, ld_b(Wh + (size_t)(nbase +  0 + n_) * HID, k0, lane), c0);
        c1 = wmma16(a, ld_b(Wh + (size_t)(nbase + 16 + n_) * HID, k0, lane), c1);
        c2 = wmma16(a, ld_b(Wh + (size_t)(nbase + 32 + n_) * HID, k0, lane), c2);
        c3 = wmma16(a, ld_b(Wh + (size_t)(nbase + 48 + n_) * HID, k0, lane), c3);
    }
    v8f acc[4] = {c0, c1, c2, c3};
    #pragma unroll
    for (int j = 0; j < 4; ++j) {
        int col = nbase + j * 16 + n_;
        int which = col >> 10, hc = col & 1023;
        int head = hc >> 6, d = hc & 63;
        float bv = bias[col];
        #pragma unroll
        for (int r = 0; r < 8; ++r) {
            int row = mbase + m0 + r;
            int b = row / NTOK, nn = row - b * NTOK;
            float v = acc[j][r] + bv;
            int bh = b * NHEADS + head;
            if (which == 0)
                Qh[((size_t)bh * NTOK + nn) * HD + d] = (_Float16)v;
            else if (which == 1)
                Kh[((size_t)bh * NTOK + nn) * HD + d] = (_Float16)v;
            else
                Vt[((size_t)bh * HD + d) * NTOK + nn] = (_Float16)v;   // pre-transposed
        }
    }
}

// ---------------- rel_h bias: relh[bh,h,w,kh] = Q[bh,hw,:]·Rh[h,kh,:] -------
// tiles: bh(64) x h(56) x mtile(2 over w) x ntile(4 over kh); 8 waves/block
__global__ void __launch_bounds__(256)
relh_kernel(const _Float16* __restrict__ Qh, const _Float16* __restrict__ Rhh,
            float* __restrict__ relh) {
    const int lane = threadIdx.x & 31, wave = threadIdx.x >> 5;
    int t = blockIdx.x * 8 + wave;
    int ntile = t & 3, mtile = (t >> 2) & 1;
    int h = (t >> 3) % Hh, bh = t / (8 * Hh);
    const int n_ = lane & 15, m0 = (lane >> 4) << 3;

    int w_a = min(mtile * 16 + (lane & 15), Ww - 1);
    const _Float16* arow = Qh + ((size_t)bh * NTOK + h * Ww + w_a) * HD;
    int kh_b = min(ntile * 16 + n_, Hh - 1);
    const _Float16* bcol = Rhh + ((size_t)h * Hh + kh_b) * HD;

    v8f c = {};
    c = wmma16(ld_a(arow, 0, lane),  ld_b(bcol, 0, lane),  c);
    c = wmma16(ld_a(arow, 32, lane), ld_b(bcol, 32, lane), c);

    #pragma unroll
    for (int r = 0; r < 8; ++r) {
        int w = mtile * 16 + m0 + r, kh = ntile * 16 + n_;
        if (w < Ww && kh < Hh)
            relh[((size_t)bh * NTOK + h * Ww + w) * Hh + kh] = c[r];
    }
}

// ---------------- rel_w bias: relw[bh,h,w,kw] = Q[bh,hw,:]·Rw[w,kw,:] -------
// tiles: bh(64) x w(28) x htile(4 over h) x kwtile(2 over kw)
__global__ void __launch_bounds__(256)
relw_kernel(const _Float16* __restrict__ Qh, const _Float16* __restrict__ Rwh,
            float* __restrict__ relw) {
    const int lane = threadIdx.x & 31, wave = threadIdx.x >> 5;
    int t = blockIdx.x * 8 + wave;
    int kwtile = t & 1, htile = (t >> 1) & 3;
    int w = (t >> 3) % Ww, bh = t / (8 * Ww);
    const int n_ = lane & 15, m0 = (lane >> 4) << 3;

    int h_a = min(htile * 16 + (lane & 15), Hh - 1);
    const _Float16* arow = Qh + ((size_t)bh * NTOK + h_a * Ww + w) * HD;
    int kw_b = min(kwtile * 16 + n_, Ww - 1);
    const _Float16* bcol = Rwh + ((size_t)w * Ww + kw_b) * HD;

    v8f c = {};
    c = wmma16(ld_a(arow, 0, lane),  ld_b(bcol, 0, lane),  c);
    c = wmma16(ld_a(arow, 32, lane), ld_b(bcol, 32, lane), c);

    #pragma unroll
    for (int r = 0; r < 8; ++r) {
        int h = htile * 16 + m0 + r, kw = kwtile * 16 + n_;
        if (h < Hh && kw < Ww)
            relw[((size_t)bh * NTOK + h * Ww + w) * Ww + kw] = c[r];
    }
}

// ---------------- fused flash attention -----------------------------------
// 1 wave per (bh, 16-query tile); 49 key tiles of 32; 4 waves/block.
__global__ void __launch_bounds__(128)
flash_kernel(const _Float16* __restrict__ Qh, const _Float16* __restrict__ Kh,
             const _Float16* __restrict__ Vt,
             const float* __restrict__ relh, const float* __restrict__ relw,
             _Float16* __restrict__ Oh) {
    __shared__ _Float16 pbuf[4][16 * 40];          // per-wave P staging, 80B rows
    const int lane = threadIdx.x & 31, wave = threadIdx.x >> 5;
    const int wid = blockIdx.x * 4 + wave;
    const int bh = wid / 98, qt = wid % 98;
    const int qbase = qt * 16;
    const int n_ = lane & 15, m0 = (lane >> 4) << 3;

    const _Float16* qrow = Qh + ((size_t)bh * NTOK + qbase + (lane & 15)) * HD;
    const v16h aQ0 = ld_a(qrow, 0, lane);
    const v16h aQ1 = ld_a(qrow, 32, lane);

    float mi[8], li[8];
    int rhb[8], rwb[8];
    #pragma unroll
    for (int r = 0; r < 8; ++r) {
        mi[r] = -INFINITY; li[r] = 0.f;
        int nq = qbase + m0 + r;
        rhb[r] = (bh * NTOK + nq) * Hh;
        rwb[r] = (bh * NTOK + nq) * Ww;
    }
    v8f o0 = {}, o1 = {}, o2 = {}, o3 = {};

    for (int kt = 0; kt < 49; ++kt) {
        const int kb = kt * 32;
        const _Float16* kcol0 = Kh + ((size_t)bh * NTOK + kb + n_) * HD;
        const _Float16* kcol1 = kcol0 + 16 * HD;
        v8f s0 = {}, s1 = {};
        s0 = wmma16(aQ0, ld_b(kcol0, 0, lane),  s0);
        s0 = wmma16(aQ1, ld_b(kcol0, 32, lane), s0);
        s1 = wmma16(aQ0, ld_b(kcol1, 0, lane),  s1);
        s1 = wmma16(aQ1, ld_b(kcol1, 32, lane), s1);

        const int j0 = kb + n_, j1 = j0 + 16;
        const int kh0 = j0 / Ww, kw0 = j0 - kh0 * Ww;
        const int kh1 = j1 / Ww, kw1 = j1 - kh1 * Ww;

        float p0[8], p1[8], corr[8];
        #pragma unroll
        for (int r = 0; r < 8; ++r) {
            float v0 = s0[r] * SCALE + relh[rhb[r] + kh0] + relw[rwb[r] + kw0];
            float v1 = s1[r] * SCALE + relh[rhb[r] + kh1] + relw[rwb[r] + kw1];
            float tmax = grp16_max(fmaxf(v0, v1));
            float mn = fmaxf(mi[r], tmax);
            float cr = __expf(mi[r] - mn);
            mi[r] = mn; corr[r] = cr;
            float e0 = __expf(v0 - mn), e1 = __expf(v1 - mn);
            p0[r] = e0; p1[r] = e1;
            li[r] = li[r] * cr + grp16_sum(e0 + e1);
        }
        #pragma unroll
        for (int r = 0; r < 8; ++r) {
            o0[r] *= corr[r]; o1[r] *= corr[r];
            o2[r] *= corr[r]; o3[r] *= corr[r];
        }
        // C-layout -> A-layout re-fragmentation of P through LDS
        #pragma unroll
        for (int r = 0; r < 8; ++r) {
            pbuf[wave][(m0 + r) * 40 + n_]      = (_Float16)p0[r];
            pbuf[wave][(m0 + r) * 40 + 16 + n_] = (_Float16)p1[r];
        }
        asm volatile("s_wait_dscnt 0" ::: "memory");
        {
            int prow = lane & 15, pk = (lane >> 4) << 3;
            union { v16h v; v8h h[2]; } up;
            up.h[0] = *(const v8h*)&pbuf[wave][prow * 40 + pk];
            up.h[1] = *(const v8h*)&pbuf[wave][prow * 40 + pk + 16];
            v16h aP = up.v;
            const _Float16* vb = Vt + ((size_t)bh * HD + n_) * NTOK + kb;
            o0 = wmma16(aP, ld_b(vb + (size_t) 0 * NTOK, 0, lane), o0);
            o1 = wmma16(aP, ld_b(vb + (size_t)16 * NTOK, 0, lane), o1);
            o2 = wmma16(aP, ld_b(vb + (size_t)32 * NTOK, 0, lane), o2);
            o3 = wmma16(aP, ld_b(vb + (size_t)48 * NTOK, 0, lane), o3);
        }
    }
    // write O in [B*N, 1024] (col = head*64+d) layout for the proj GEMM
    const int b = bh >> 4, head = bh & 15;
    #pragma unroll
    for (int r = 0; r < 8; ++r) {
        float inv = 1.0f / li[r];
        size_t ro = ((size_t)b * NTOK + qbase + m0 + r) * HID + head * HD;
        Oh[ro +  0 + n_] = (_Float16)(o0[r] * inv);
        Oh[ro + 16 + n_] = (_Float16)(o1[r] * inv);
        Oh[ro + 32 + n_] = (_Float16)(o2[r] * inv);
        Oh[ro + 48 + n_] = (_Float16)(o3[r] * inv);
    }
}

// ---------------- output projection: [6272,1024]x[1024,1024]^T + bias ------
__global__ void __launch_bounds__(256)
proj_gemm_kernel(const _Float16* __restrict__ Oh, const _Float16* __restrict__ Ph,
                 const float* __restrict__ bias, float* __restrict__ out) {
    const int lane = threadIdx.x & 31, wave = threadIdx.x >> 5;
    const int mbase = blockIdx.x * 64 + (wave & 3) * 16;
    const int nbase = blockIdx.y * 128 + (wave >> 2) * 64;
    const int n_ = lane & 15, m0 = (lane >> 4) << 3;

    const _Float16* arow = Oh + (size_t)(mbase + (lane & 15)) * HID;
    v8f c0 = {}, c1 = {}, c2 = {}, c3 = {};
    for (int k0 = 0; k0 < HID; k0 += 32) {
        v16h a = ld_a(arow, k0, lane);
        c0 = wmma16(a, ld_b(Ph + (size_t)(nbase +  0 + n_) * HID, k0, lane), c0);
        c1 = wmma16(a, ld_b(Ph + (size_t)(nbase + 16 + n_) * HID, k0, lane), c1);
        c2 = wmma16(a, ld_b(Ph + (size_t)(nbase + 32 + n_) * HID, k0, lane), c2);
        c3 = wmma16(a, ld_b(Ph + (size_t)(nbase + 48 + n_) * HID, k0, lane), c3);
    }
    v8f acc[4] = {c0, c1, c2, c3};
    #pragma unroll
    for (int j = 0; j < 4; ++j) {
        int col = nbase + j * 16 + n_;
        float bv = bias[col];
        #pragma unroll
        for (int r = 0; r < 8; ++r)
            out[(size_t)(mbase + m0 + r) * HID + col] = acc[j][r] + bv;
    }
}

// ---------------- host launcher ----------------
extern "C" void kernel_launch(void* const* d_in, const int* in_sizes, int n_in,
                              void* d_out, int out_size, void* d_ws, size_t ws_size,
                              hipStream_t stream) {
    const float* hid     = (const float*)d_in[0];
    const float* qkv_w   = (const float*)d_in[1];
    const float* qkv_b   = (const float*)d_in[2];
    const float* proj_w  = (const float*)d_in[3];
    const float* proj_b  = (const float*)d_in[4];
    const float* rel_h_t = (const float*)d_in[5];
    const float* rel_w_t = (const float*)d_in[6];

    char* p = (char*)d_ws;
    auto carve = [&](size_t bytes) -> void* {
        void* r = (void*)p;
        p += (bytes + 255) & ~(size_t)255;
        return r;
    };
    _Float16* Xh  = (_Float16*)carve((size_t)ROWS * HID * 2);
    _Float16* Wqh = (_Float16*)carve((size_t)3 * HID * HID * 2);
    _Float16* Pwh = (_Float16*)carve((size_t)HID * HID * 2);
    _Float16* Qh  = (_Float16*)carve((size_t)BH * NTOK * HD * 2);
    _Float16* Kh  = (_Float16*)carve((size_t)BH * NTOK * HD * 2);
    _Float16* Vt  = (_Float16*)carve((size_t)BH * NTOK * HD * 2);
    _Float16* Rhh = (_Float16*)carve((size_t)Hh * Hh * HD * 2);
    _Float16* Rwh = (_Float16*)carve((size_t)Ww * Ww * HD * 2);
    float*    relh = (float*)carve((size_t)BH * NTOK * Hh * 4);
    float*    relw = (float*)carve((size_t)BH * NTOK * Ww * 4);
    _Float16* Oh  = (_Float16*)carve((size_t)ROWS * HID * 2);

    cast_f32_f16_kernel<<<2048, 256, 0, stream>>>(hid, Xh, ROWS * HID);
    cast_f32_f16_kernel<<<2048, 256, 0, stream>>>(qkv_w, Wqh, 3 * HID * HID);
    cast_f32_f16_kernel<<<1024, 256, 0, stream>>>(proj_w, Pwh, HID * HID);
    gather_rel_kernel<<<512, 256, 0, stream>>>(rel_h_t, Rhh, Hh);
    gather_rel_kernel<<<256, 256, 0, stream>>>(rel_w_t, Rwh, Ww);

    qkv_gemm_kernel<<<dim3(ROWS / 64, 3 * HID / 128), 256, 0, stream>>>(
        Xh, Wqh, qkv_b, Qh, Kh, Vt);

    relh_kernel<<<(BH * Hh * 8) / 8, 256, 0, stream>>>(Qh, Rhh, relh);
    relw_kernel<<<(BH * Ww * 8) / 8, 256, 0, stream>>>(Qh, Rwh, relw);

    flash_kernel<<<(BH * (NTOK / 16)) / 4, 128, 0, stream>>>(
        Qh, Kh, Vt, relh, relw, Oh);

    proj_gemm_kernel<<<dim3(ROWS / 64, HID / 128), 256, 0, stream>>>(
        Oh, Pwh, proj_b, (float*)d_out);
}
